// LayoutLMv2SelfAttention_10977936409147
// MI455X (gfx1250) — matrix operand order repaired and checked by reading, not compiled, not run on hardware
//
#include <hip/hip_runtime.h>

// LayoutLMv2 self-attention forward for MI455X (gfx1250, wave32, WMMA).
// B=8, S=1024, H=12, DH=64, HID=768.
// Outputs (concat): context [B,S,768] fp32, probs [B,H,S,S] fp32.

#define BATCH 8
#define SEQ   1024
#define HEADS 12
#define DHEAD 64
#define HID   768
#define QKV3  (3 * HID)     // 2304

typedef __attribute__((ext_vector_type(16))) __bf16 bf16x16;
typedef __attribute__((ext_vector_type(8)))  float  f32x8;

union FragAB {
    bf16x16      v;
    unsigned int d[8];
};

// fp32 -> bf16 bits, round-to-nearest-even
static __device__ __forceinline__ unsigned short f2bf(float f) {
    unsigned int u = __float_as_uint(f);
    u += 0x7FFFu + ((u >> 16) & 1u);
    return (unsigned short)(u >> 16);
}

// ---- 16x32 bf16 A fragment (ISA 7.12.2: lanes 0-15 rows M=0..15, K groups
// interleaved by 8 between lane halves; VGPR v<4 -> K=2v, v>=4 -> K=2v+8) ----
static __device__ __forceinline__ void load_fragA(FragAB& f, const unsigned short* p,
                                                  int ld, int row0, int k0) {
    int lane = threadIdx.x & 31;
    int m = lane & 15, g = lane >> 4;
    const unsigned short* rp = p + (size_t)(row0 + m) * ld + k0 + 8 * g;
#pragma unroll
    for (int v = 0; v < 8; ++v) {
        int k = (v < 4) ? 2 * v : 2 * v + 8;
        f.d[v] = *(const unsigned int*)(rp + k);
    }
}

// ---- 32x16 bf16 B fragment: lane n holds column n; K groups interleaved by 8
// between lane halves (same as A); VGPR v<4 -> K=2v+8g, v>=4 -> K=2v+8+8g. Source stored column-major
// (i.e. rows of the transposed matrix) so pairs are contiguous. ----
static __device__ __forceinline__ void load_fragB(FragAB& f, const unsigned short* p,
                                                  int ld, int col0, int k0) {
    int lane = threadIdx.x & 31;
    int n = lane & 15, g = lane >> 4;
    const unsigned short* rp = p + (size_t)(col0 + n) * ld + k0 + 8 * g;
#pragma unroll
    for (int v = 0; v < 8; ++v) {
        int k = (v < 4) ? 2 * v : 2 * v + 8;
        f.d[v] = *(const unsigned int*)(rp + k);
    }
}

// A fragment sourced from fp32 (probs), converted to bf16 on the fly.
static __device__ __forceinline__ void load_fragA_f32(FragAB& f, const float* p,
                                                      int ld, int row0, int k0) {
    int lane = threadIdx.x & 31;
    int m = lane & 15, g = lane >> 4;
    const float* rp = p + (size_t)(row0 + m) * ld + k0 + 8 * g;
#pragma unroll
    for (int v = 0; v < 8; ++v) {
        int k = (v < 4) ? 2 * v : 2 * v + 8;
        unsigned int lo = f2bf(rp[k]);
        unsigned int hi = f2bf(rp[k + 1]);
        f.d[v] = lo | (hi << 16);
    }
}

static __device__ __forceinline__ f32x8 wmma_bf16(const FragAB& a, const FragAB& b, f32x8 c) {
    return __builtin_amdgcn_wmma_f32_16x16x32_bf16(false, a.v, false, b.v,
                                                   (short)0, c, false, false);
}

// ---------------- K0a: X fp32 -> bf16 ----------------
__global__ __launch_bounds__(256) void cvt_x_kernel(const float* __restrict__ X,
                                                    unsigned short* __restrict__ Xb, int n) {
    int i = blockIdx.x * 256 + threadIdx.x;
    if (i < n) Xb[i] = f2bf(X[i]);
}

// ---------------- K0b: W [768,2304] fp32 -> Wt [2304,768] bf16 ----------------
__global__ __launch_bounds__(256) void cvt_wt_kernel(const float* __restrict__ W,
                                                     unsigned short* __restrict__ Wt) {
    int i = blockIdx.x * 256 + threadIdx.x;
    if (i < QKV3 * HID) {
        int n = i / HID, k = i - n * HID;
        Wt[i] = f2bf(W[(size_t)k * QKV3 + n]);
    }
}

// ---------------- K1: QKV GEMM (bf16 WMMA, K=768) ----------------
// One 16x16 output tile per wave; fuses q_bias + 1/sqrt(DH), v_bias.
// Writes Q,K as [B,H,S,64] bf16 and V transposed as [B,H,64,S] bf16.
__global__ __launch_bounds__(128) void qkv_gemm_kernel(
    const unsigned short* __restrict__ Xb, const unsigned short* __restrict__ Wt,
    const float* __restrict__ qb, const float* __restrict__ vb,
    unsigned short* __restrict__ Qw, unsigned short* __restrict__ Kw,
    unsigned short* __restrict__ Vt) {
    const int NT = QKV3 / 16;  // 144 column tiles
    int wave = blockIdx.x * 4 + (threadIdx.x >> 5);
    int mt = wave / NT, nt = wave - mt * NT;
    int row0 = mt * 16, col0 = nt * 16;

    f32x8 acc = {};
    FragAB a, bf;
#pragma unroll 4
    for (int k0 = 0; k0 < HID; k0 += 32) {
        load_fragA(a, Xb, HID, row0, k0);
        load_fragB(bf, Wt, HID, col0, k0);
        acc = wmma_bf16(a, bf, acc);
    }

    int lane = threadIdx.x & 31;
    int n = lane & 15, g = lane >> 4;
    int col = col0 + n;
    int seg = col / HID;      // 0=Q 1=K 2=V (uniform per wave: 768 % 16 == 0)
    int wcol = col - seg * HID;
    int h = wcol >> 6, dh = wcol & 63;
    int b = row0 >> 10;
    int sbase = row0 & (SEQ - 1);
    size_t head = (size_t)(b * HEADS + h);
#pragma unroll
    for (int r = 0; r < 8; ++r) {
        int s = sbase + r + 8 * g;
        float val = acc[r];
        if (seg == 0) {
            val = (val + qb[wcol]) * 0.125f;  // + q_bias, * 1/sqrt(64)
            Qw[(head * SEQ + s) * DHEAD + dh] = f2bf(val);
        } else if (seg == 1) {
            Kw[(head * SEQ + s) * DHEAD + dh] = f2bf(val);
        } else {
            val += vb[wcol];
            Vt[(head * DHEAD + dh) * SEQ + s] = f2bf(val);  // transposed for PV B-frags
        }
    }
}

// ---------------- K2: scores = QK^T + rel + rel2d, mask, softmax ----------------
// One block (4 waves) per 16-row query tile of one (b,h). Score strip lives in
// 64KB LDS; rel tensors are streamed with non-temporal loads (read-once data,
// keep L2 for Q/K/V + probs); softmax via wave32 shuffles with fully coalesced
// 128B probs stores.
__global__ __launch_bounds__(128) void scores_softmax_kernel(
    const unsigned short* __restrict__ Qw, const unsigned short* __restrict__ Kw,
    const float* __restrict__ rel, const float* __restrict__ rel2d,
    const unsigned char* __restrict__ amask,  // bool mask [B,1,1,S], nonzero = masked
    float* __restrict__ probs) {
    __shared__ float sc[16 * SEQ];  // 64 KB (WGP has 320 KB)

    int tile = blockIdx.x;
    int qt = tile & 63, bh = tile >> 6;  // 64 q-tiles per head-batch
    int b = bh / HEADS;
    int q0 = qt * 16;
    int wv = threadIdx.x >> 5, lane = threadIdx.x & 31;
    int n = lane & 15, g = lane >> 4;

    const unsigned short* Qh = Qw + (size_t)bh * SEQ * DHEAD;
    const unsigned short* Kh = Kw + (size_t)bh * SEQ * DHEAD;
    const float* relh  = rel   + (size_t)bh * SEQ * SEQ;
    const float* rel2h = rel2d + (size_t)bh * SEQ * SEQ;

    FragAB a0, a1;
    load_fragA(a0, Qh, DHEAD, q0, 0);
    load_fragA(a1, Qh, DHEAD, q0, 32);

    for (int nt = wv; nt < SEQ / 16; nt += 4) {
        FragAB b0, b1;
        load_fragB(b0, Kh, DHEAD, nt * 16, 0);   // B(k=dh, n=s_col) = K[s_col][dh]
        load_fragB(b1, Kh, DHEAD, nt * 16, 32);
        f32x8 acc = {};
        acc = wmma_bf16(a0, b0, acc);
        acc = wmma_bf16(a1, b1, acc);

        int c = nt * 16 + n;
        bool masked = amask[b * SEQ + c] != 0;
#pragma unroll
        for (int r = 0; r < 8; ++r) {
            int m = r + 8 * g;
            size_t ri = (size_t)(q0 + m) * SEQ + c;
            // stream-once data: non-temporal so it doesn't evict Q/K/V/probs in L2
            float v = acc[r] + __builtin_nontemporal_load(&relh[ri])
                             + __builtin_nontemporal_load(&rel2h[ri]);
            if (masked) v = -1e30f;
            sc[m * SEQ + c] = v;
        }
    }
    __syncthreads();

    // Softmax: each wave owns 4 rows; 32 lanes sweep contiguous columns
    // (conflict-free LDS, fully coalesced 128B global stores).
#pragma unroll
    for (int rr = 0; rr < 4; ++rr) {
        int r = wv * 4 + rr;
        float* srow = &sc[r * SEQ];

        float mx = -3.4e38f;
        for (int c = lane; c < SEQ; c += 32) mx = fmaxf(mx, srow[c]);
#pragma unroll
        for (int o = 1; o < 32; o <<= 1) mx = fmaxf(mx, __shfl_xor(mx, o, 32));

        float sum = 0.f;
        for (int c = lane; c < SEQ; c += 32) {
            float e = __expf(srow[c] - mx);
            srow[c] = e;
            sum += e;
        }
#pragma unroll
        for (int o = 1; o < 32; o <<= 1) sum += __shfl_xor(sum, o, 32);
        float inv = 1.0f / sum;

        float* prow = probs + (size_t)bh * SEQ * SEQ + (size_t)(q0 + r) * SEQ;
        for (int c = lane; c < SEQ; c += 32) prow[c] = srow[c] * inv;
    }
}

// ---------------- K3: context = P @ V ----------------
// One wave per 16-row query tile; 4 accumulators cover all 64 head columns
// (A-fragment reused across the 4 N-tiles). P read fp32 -> bf16 on the fly.
__global__ __launch_bounds__(128) void context_kernel(
    const float* __restrict__ probs, const unsigned short* __restrict__ Vt,
    float* __restrict__ ctx) {
    int wave = blockIdx.x * 4 + (threadIdx.x >> 5);
    int qt = wave & 63, bh = wave >> 6;
    int b = bh / HEADS, h = bh - b * HEADS;
    int q0 = qt * 16;

    const float* Ph = probs + (size_t)bh * SEQ * SEQ;
    const unsigned short* Vh = Vt + (size_t)bh * DHEAD * SEQ;

    f32x8 acc0 = {}, acc1 = {}, acc2 = {}, acc3 = {};
    for (int k0 = 0; k0 < SEQ; k0 += 32) {
        FragAB a;
        load_fragA_f32(a, Ph, SEQ, q0, k0);
        FragAB bf;
        load_fragB(bf, Vh, SEQ, 0, k0);  acc0 = wmma_bf16(a, bf, acc0);
        load_fragB(bf, Vh, SEQ, 16, k0); acc1 = wmma_bf16(a, bf, acc1);
        load_fragB(bf, Vh, SEQ, 32, k0); acc2 = wmma_bf16(a, bf, acc2);
        load_fragB(bf, Vh, SEQ, 48, k0); acc3 = wmma_bf16(a, bf, acc3);
    }

    int lane = threadIdx.x & 31;
    int n = lane & 15, g = lane >> 4;
#pragma unroll
    for (int r = 0; r < 8; ++r) {
        int s = q0 + r + 8 * g;
        float* orow = ctx + ((size_t)b * SEQ + s) * HID + h * DHEAD;
        orow[n]      = acc0[r];
        orow[16 + n] = acc1[r];
        orow[32 + n] = acc2[r];
        orow[48 + n] = acc3[r];
    }
}

extern "C" void kernel_launch(void* const* d_in, const int* in_sizes, int n_in,
                              void* d_out, int out_size, void* d_ws, size_t ws_size,
                              hipStream_t stream) {
    const float*         X     = (const float*)d_in[0];          // [8,1024,768]
    const unsigned char* amask = (const unsigned char*)d_in[1];  // bool [8,1,1,1024]
    const float*         rel   = (const float*)d_in[2];          // [8,12,1024,1024]
    const float*         rel2  = (const float*)d_in[3];          // [8,12,1024,1024]
    const float*         W     = (const float*)d_in[4];          // [768,2304]
    const float*         qb    = (const float*)d_in[5];          // [768]
    const float*         vb    = (const float*)d_in[6];          // [768]

    float* ctx   = (float*)d_out;                                // [8,1024,768]
    float* probs = ctx + (size_t)BATCH * SEQ * HID;              // [8,12,1024,1024]

    // Workspace layout (bf16 = ushort), total ~54 MB.
    unsigned short* Xb = (unsigned short*)d_ws;                  // [8192,768]
    unsigned short* Wt = Xb + (size_t)BATCH * SEQ * HID;         // [2304,768]
    unsigned short* Qw = Wt + (size_t)QKV3 * HID;                // [B,H,S,64]
    unsigned short* Kw = Qw + (size_t)BATCH * HEADS * SEQ * DHEAD;
    unsigned short* Vt = Kw + (size_t)BATCH * HEADS * SEQ * DHEAD;  // [B,H,64,S]

    const int nX = BATCH * SEQ * HID;
    cvt_x_kernel<<<(nX + 255) / 256, 256, 0, stream>>>(X, Xb, nX);
    cvt_wt_kernel<<<(QKV3 * HID + 255) / 256, 256, 0, stream>>>(W, Wt);

    // 512 row-tiles x 144 col-tiles, 4 waves (tiles) per 128-thread block.
    qkv_gemm_kernel<<<(512 * 144) / 4, 128, 0, stream>>>(Xb, Wt, qb, vb, Qw, Kw, Vt);

    // One block per (b,h,16-row q tile): 8*12*64 = 6144 blocks.
    scores_softmax_kernel<<<BATCH * HEADS * (SEQ / 16), 128, 0, stream>>>(
        Qw, Kw, rel, rel2, amask, probs);

    // 6144 query tiles / 4 waves per block.
    context_kernel<<<BATCH * HEADS * (SEQ / 16) / 4, 128, 0, stream>>>(probs, Vt, ctx);
}
